// GaussianUpsampling_34961033789489
// MI455X (gfx1250) — matrix operand order repaired and checked by reading, not compiled, not run on hardware
//
#include <hip/hip_runtime.h>
#include <stdint.h>

typedef __attribute__((ext_vector_type(16))) _Float16 v16h;
typedef __attribute__((ext_vector_type(8)))  _Float16 v8h;
typedef __attribute__((ext_vector_type(4)))  _Float16 v4h;
typedef __attribute__((ext_vector_type(8)))  float    v8f;

#define Bc 8
#define Nc 512
#define Dc 768
#define Tc 4096

#define DSEG 96                 // d columns handled per block
#define TBLK 512                // t frames handled per block
#define MSUB 64                 // t frames per inner subtile
#define NSUB (TBLK / MSUB)      // 8 subtiles

#define SE_ROWP 104             // enc panel LDS row stride in halves (96 + 8 pad, 16B-aligned)
#define SA_ROWP 24              // A panel LDS row stride in halves (16 + 8 pad, 16B-aligned)

constexpr int SE_BYTES  = Nc * SE_ROWP * 2;            // 106496
constexpr int SA_BYTES  = 4 * 16 * 32 * SA_ROWP * 2;   // 98304
constexpr int SP_BYTES  = Nc * 4 * 3;                  // sC, sInv, sCoef
constexpr int SS_BYTES  = 64 * 4 * 2;                  // sS, srcp
constexpr int SMEM_BYTES = SE_BYTES + SA_BYTES + SP_BYTES + SS_BYTES;  // ~207 KB

// ---- CDNA5 async global->LDS copy (ASYNCcnt) ------------------------------
__device__ __forceinline__ void async_b128(const void* gsrc, uint32_t lds_addr) {
    asm volatile("global_load_async_to_lds_b128 %0, %1, off"
                 :: "v"(lds_addr), "v"(gsrc) : "memory");
}
__device__ __forceinline__ void wait_async0() {
    asm volatile("s_wait_asynccnt 0x0" ::: "memory");
}

// ---------------------------------------------------------------------------
// Kernel 1: per-batch serial cumsum -> Gaussian centers c[b,n]. Trivial size.
// ---------------------------------------------------------------------------
__global__ void centers_kernel(const int* __restrict__ dur, float* __restrict__ cbuf) {
    int b = threadIdx.x;
    if (b >= Bc) return;
    const int* d = dur + b * Nc;
    float* cb = cbuf + b * Nc;
    float run = 0.0f;
    for (int n = 0; n < Nc; ++n) {
        float df = (float)d[n];
        run += df;
        cb[n] = run - 0.5f * df;
    }
}

// ---------------------------------------------------------------------------
// Kernel 2: one-time f32 -> f16 conversion of encoder outputs (6.3 MB).
// ---------------------------------------------------------------------------
__global__ void cvt_f16_kernel(const float* __restrict__ src, _Float16* __restrict__ dst) {
    int i = blockIdx.x * 256 + threadIdx.x;           // one float4 per thread
    float4 f = ((const float4*)src)[i];
    v4h h;
    h[0] = (_Float16)f.x; h[1] = (_Float16)f.y;
    h[2] = (_Float16)f.z; h[3] = (_Float16)f.w;
    ((v4h*)dst)[i] = h;
}

// ---------------------------------------------------------------------------
// Kernel 3 (main): block = (t-group of 512, d-slice of 96, batch).
//   stage:  enc f16 panel (512 x 96) -> LDS via global_load_async_to_lds_b128
//   per 64-t subtile:
//     B1: Gaussian w -> A panel in LDS (f16, WMMA fragment order) + S[t] sums
//     B2: attn = w * rcp(S)  (recomputed exactly in f32; d-slice 0 only)
//     B3: out tile = rcp(S) * (w^T @ enc) via v_wmma_f32_16x16x32_f16
// 256 threads = 8 waves; wave = (mt 0..3, d-half 0..1), 3 acc tiles each.
// ---------------------------------------------------------------------------
__launch_bounds__(256)
__global__ void gauss_attn_kernel(const _Float16* __restrict__ ench,
                                  const float* __restrict__ rng,
                                  const uint8_t* __restrict__ mask,
                                  const float* __restrict__ cbuf,
                                  float* __restrict__ out,     // [B, T, D]
                                  float* __restrict__ attn) {  // [B, N, T]
    extern __shared__ char smem[];
    _Float16* sE   = (_Float16*)smem;                          // [Nc][SE_ROWP]
    _Float16* sA   = (_Float16*)(smem + SE_BYTES);             // [4*16*32][SA_ROWP]
    float*    sC   = (float*)(smem + SE_BYTES + SA_BYTES);     // [Nc] centers
    float*    sInv = sC + Nc;                                  // [Nc] 1/s
    float*    sCoef= sInv + Nc;                                // [Nc] coef (0 if masked)
    float*    sS   = sCoef + Nc;                               // [64]
    float*    srcp = sS + 64;                                  // [64]

    const int tid = threadIdx.x;
    const int b   = blockIdx.z;
    const int d0  = blockIdx.y * DSEG;
    const int tg0 = blockIdx.x * TBLK;
    const bool do_attn = (blockIdx.y == 0);

    // ---- per-n parameters into LDS ----
    for (int n = tid; n < Nc; n += 256) {
        float s = rng[b * Nc + n];
        float inv = 1.0f / s;
        sC[n]    = cbuf[b * Nc + n];
        sInv[n]  = inv;
        sCoef[n] = mask[b * Nc + n] ? 0.0f : 0.3989422804014327f * inv;
    }

    // ---- stage enc f16 panel via async copies (24 x b128 per thread) ----
    {
        const char* gbase = (const char*)(ench + ((size_t)b * Nc * Dc + d0));
        uint32_t lbase = (uint32_t)(uintptr_t)(void*)sE;
#pragma unroll
        for (int j = 0; j < 24; ++j) {
            int seg = j * 256 + tid;        // 6144 segments of 16B (12 per row)
            int row = seg / 12;
            int c   = seg - row * 12;
            async_b128(gbase + (size_t)row * (Dc * 2) + c * 16,
                       lbase + (uint32_t)(row * (SE_ROWP * 2) + c * 16));
        }
    }
    wait_async0();
    __syncthreads();

    const int lane = tid & 31;
    const int wv   = tid >> 5;
    const int mt   = wv >> 1;          // which 16-row M tile of the 64-t subtile
    const int dh   = wv & 1;           // which 48-column d half of the 96 slice
    const int tl   = tid & 63;         // t within subtile (phase B mapping)
    const int rep  = tid >> 6;         // 4 n-range groups of 128

    for (int st = 0; st < NSUB; ++st) {
        const int t0 = tg0 + st * MSUB;

        if (tid < 64) sS[tid] = 0.0f;
        __syncthreads();

        // ---- B1: Gaussian weights -> f16 A panel (fragment order) + sums ----
        {
            const float tv = (float)(t0 + tl + 1);   // t = 1..T
            const int m  = tl & 15;
            const int mtw = tl >> 4;
            float part = 0.0f;
            for (int n = rep * 128; n < rep * 128 + 128; ++n) {
                float z = (tv - sC[n]) * sInv[n];
                float w = sCoef[n] * __expf(-0.5f * z * z);
                part += w;
                int kloc  = n & 31;
                int ci    = n >> 5;
                int khalf = (kloc >> 3) & 1;
                int h     = (kloc & 7) + 8 * ((kloc >> 4) & 1);
                int l     = m + 16 * khalf;
                sA[(((mtw * 16 + ci) * 32) + l) * SA_ROWP + h] = (_Float16)w;
            }
            atomicAdd(&sS[tl], part);
        }
        __syncthreads();
        if (tid < 64) srcp[tid] = 1.0f / (sS[tid] + 1e-8f);
        __syncthreads();

        // ---- B2: exact f32 attention (recompute exp; d-slice 0 only) ----
        if (do_attn) {
            const float tv = (float)(t0 + tl + 1);
            const float rc = srcp[tl];
            for (int n = rep * 128; n < rep * 128 + 128; ++n) {
                float z = (tv - sC[n]) * sInv[n];
                float w = sCoef[n] * __expf(-0.5f * z * z);
                attn[((size_t)(b * Nc + n)) * Tc + (t0 + tl)] = w * rc;
            }
        }

        // ---- B3: WMMA GEMM, A and B both straight from LDS as f16 ----
        v8f acc[3] = {};
        for (int ci = 0; ci < 16; ++ci) {
            const _Float16* ap = sA + (((mt * 16 + ci) * 32) + lane) * SA_ROWP;
            v8h alo = *(const v8h*)ap;
            v8h ahi = *(const v8h*)(ap + 8);
            v16h af = __builtin_shufflevector(alo, ahi, 0,1,2,3,4,5,6,7,8,9,10,11,12,13,14,15);
            const _Float16* bp = sE + (ci * 32) * SE_ROWP + dh * 48 + (lane & 15);
#pragma unroll
            for (int dt = 0; dt < 3; ++dt) {
                v16h bf;
#pragma unroll
                for (int s = 0; s < 16; ++s)
                    bf[s] = bp[((s & 8) * 2 + 8 * (lane >> 4) + (s & 7)) * SE_ROWP + dt * 16];
                acc[dt] = __builtin_amdgcn_wmma_f32_16x16x32_f16(
                    false, af, false, bf, (short)0, acc[dt], false, false);
            }
        }

        // ---- epilogue: scale by 1/(S+eps), store f32 ----
        {
            const int hi8  = (lane < 16) ? 0 : 8;
            const int dcol = lane & 15;
            const int tb   = t0 + mt * 16 + hi8;
#pragma unroll
            for (int dt = 0; dt < 3; ++dt) {
#pragma unroll
                for (int r = 0; r < 8; ++r) {
                    out[((size_t)(b * Tc + tb + r)) * Dc + (d0 + dh * 48 + dt * 16 + dcol)] =
                        acc[dt][r] * srcp[mt * 16 + hi8 + r];
                }
            }
        }
        __syncthreads();   // protect sA/sS before next subtile
    }
}

// ---------------------------------------------------------------------------
extern "C" void kernel_launch(void* const* d_in, const int* in_sizes, int n_in,
                              void* d_out, int out_size, void* d_ws, size_t ws_size,
                              hipStream_t stream) {
    const float*   enc  = (const float*)d_in[0];    // [B,N,D] f32
    const int*     dur  = (const int*)d_in[1];      // [B,N] i32
    const float*   rng  = (const float*)d_in[2];    // [B,N] f32
    const uint8_t* mask = (const uint8_t*)d_in[3];  // [B,N] bool
    (void)in_sizes; (void)n_in; (void)out_size; (void)ws_size;

    float* out  = (float*)d_out;                       // [B,T,D]
    float* attn = out + (size_t)Bc * Tc * Dc;          // [B,N,T]

    float*     cbuf = (float*)d_ws;                              // B*N*4 = 16 KB
    _Float16*  ench = (_Float16*)((char*)d_ws + Bc * Nc * 4);    // B*N*D*2 = 6.3 MB

    (void)hipFuncSetAttribute((const void*)gauss_attn_kernel,
                              hipFuncAttributeMaxDynamicSharedMemorySize, SMEM_BYTES);

    centers_kernel<<<1, 64, 0, stream>>>(dur, cbuf);
    cvt_f16_kernel<<<(Bc * Nc * Dc / 4) / 256, 256, 0, stream>>>(enc, ench);

    dim3 grid(Tc / TBLK, Dc / DSEG, Bc);
    gauss_attn_kernel<<<grid, 256, SMEM_BYTES, stream>>>(ench, rng, mask, cbuf, out, attn);
}
